// DeepseekMoE_2585570312833
// MI455X (gfx1250) — matrix operand order repaired and checked by reading, not compiled, not run on hardware
//
#include <hip/hip_runtime.h>
#include <cstdint>
#include <cstddef>

// ---------------- problem constants (match reference) ----------------
#define HDIM 2048
#define NEXP 16
#define KTOP 4
#define IDIM 1408
#define ISDIM 2816
#define TTOK 1024
#define LDSW 40   // padded LDS row length (bf16): 80 B rows, 16 B aligned

typedef __attribute__((ext_vector_type(16))) __bf16 v16bf;
typedef __attribute__((ext_vector_type(8)))  __bf16 v8bf;
typedef __attribute__((ext_vector_type(4)))  __bf16 v4bf;
typedef __attribute__((ext_vector_type(8)))  float  v8f;

__device__ __forceinline__ v8f bwmma(v16bf a, v16bf b, v8f c) {
  return __builtin_amdgcn_wmma_f32_16x16x32_bf16(
      /*neg_a=*/false, a, /*neg_b=*/false, b,
      /*c_mod=*/(short)0, c, /*reuse_a=*/false, /*reuse_b=*/false);
}

// build a 16-element bf16 fragment from two 16-byte LDS chunks (2x ds_load_b128)
__device__ __forceinline__ v16bf frag16(const __bf16* p0, const __bf16* p1) {
  const v8bf lo = *reinterpret_cast<const v8bf*>(p0);
  const v8bf hi = *reinterpret_cast<const v8bf*>(p1);
  return __builtin_shufflevector(lo, hi,
      0,1,2,3,4,5,6,7,8,9,10,11,12,13,14,15);
}

// Load one 32(K) x 128(N) fp32 weight tile, convert to bf16, store transposed
// [n][k] into LDS with packed b64 stores. Thread owns columns [wn,wn+4) and
// k-rows [8*kq, 8*kq+8). Global loads are fully coalesced float4.
__device__ __forceinline__ void load_w_tile(const float* __restrict__ Wt,
                                            size_t ld, int wn, int kq,
                                            __bf16 (*lds)[LDSW]) {
  const float* base = Wt + (size_t)(8 * kq) * ld + wn;
#pragma unroll
  for (int kh = 0; kh < 8; kh += 4) {
    const float4 r0 = *reinterpret_cast<const float4*>(base + (size_t)(kh + 0) * ld);
    const float4 r1 = *reinterpret_cast<const float4*>(base + (size_t)(kh + 1) * ld);
    const float4 r2 = *reinterpret_cast<const float4*>(base + (size_t)(kh + 2) * ld);
    const float4 r3 = *reinterpret_cast<const float4*>(base + (size_t)(kh + 3) * ld);
    const int k0 = 8 * kq + kh;
    const v4bf c0 = {(__bf16)r0.x, (__bf16)r1.x, (__bf16)r2.x, (__bf16)r3.x};
    const v4bf c1 = {(__bf16)r0.y, (__bf16)r1.y, (__bf16)r2.y, (__bf16)r3.y};
    const v4bf c2 = {(__bf16)r0.z, (__bf16)r1.z, (__bf16)r2.z, (__bf16)r3.z};
    const v4bf c3 = {(__bf16)r0.w, (__bf16)r1.w, (__bf16)r2.w, (__bf16)r3.w};
    *reinterpret_cast<v4bf*>(&lds[wn + 0][k0]) = c0;
    *reinterpret_cast<v4bf*>(&lds[wn + 1][k0]) = c1;
    *reinterpret_cast<v4bf*>(&lds[wn + 2][k0]) = c2;
    *reinterpret_cast<v4bf*>(&lds[wn + 3][k0]) = c3;
  }
}

// ====================================================================
// Kernel 1: router. One wave per token: 16 logits, softmax, top-4,
// renormalize, atomic scatter into per-expert gather lists.
// ====================================================================
__global__ __launch_bounds__(256)
void moe_router_k(const float* __restrict__ x, const float* __restrict__ gw,
                  int* __restrict__ counts, int* __restrict__ tlist,
                  float* __restrict__ wlist,
                  int* __restrict__ s_tlist, float* __restrict__ s_wlist) {
  const int lane = threadIdx.x & 31;
  const int wv   = threadIdx.x >> 5;
  const int t    = blockIdx.x * 8 + wv;
  if (t >= TTOK) return;

  float acc[NEXP];
#pragma unroll
  for (int e = 0; e < NEXP; ++e) acc[e] = 0.0f;

  for (int h = lane; h < HDIM; h += 32) {
    const float xv = x[(size_t)t * HDIM + h];
#pragma unroll
    for (int e = 0; e < NEXP; ++e) acc[e] += xv * gw[(size_t)e * HDIM + h];
  }
#pragma unroll
  for (int e = 0; e < NEXP; ++e) {
#pragma unroll
    for (int off = 16; off > 0; off >>= 1)
      acc[e] += __shfl_xor(acc[e], off, 32);
  }

  if (lane == 0) {
    float mx = acc[0];
#pragma unroll
    for (int e = 1; e < NEXP; ++e) mx = fmaxf(mx, acc[e]);
    float p[NEXP], s = 0.0f;
#pragma unroll
    for (int e = 0; e < NEXP; ++e) { p[e] = __expf(acc[e] - mx); s += p[e]; }
    const float inv = 1.0f / s;
#pragma unroll
    for (int e = 0; e < NEXP; ++e) p[e] *= inv;

    int   idx[KTOP]; float val[KTOP]; float sum4 = 0.0f;
#pragma unroll
    for (int k = 0; k < KTOP; ++k) {
      int bj = 0; float bv = -1.0f;
#pragma unroll
      for (int e = 0; e < NEXP; ++e) if (p[e] > bv) { bv = p[e]; bj = e; }
      idx[k] = bj; val[k] = bv; p[bj] = -2.0f; sum4 += bv;
    }
    const float invs = 1.0f / sum4;
#pragma unroll
    for (int k = 0; k < KTOP; ++k) {
      const int e    = idx[k];
      const int slot = atomicAdd(&counts[e], 1);
      tlist[e * TTOK + slot] = t;
      wlist[e * TTOK + slot] = val[k] * invs;
    }
    s_tlist[t] = t;
    s_wlist[t] = 1.0f;
    if (t == 0) counts[NEXP] = TTOK;  // shared-expert "count"
  }
}

// ====================================================================
// Kernel 2: gate+up GEMM + SiLU epilogue. Tile = 32 tokens x 128 cols,
// K-step 32. 128 threads / 4 waves; 8 wmma per wave per K-step
// (B fragments reused across the two 16-row M subtiles).
// ====================================================================
__global__ __launch_bounds__(128)
void moe_gateup_k(const float* __restrict__ x,
                  const float* __restrict__ Wg, const float* __restrict__ Wu,
                  const int* __restrict__ tlist, const float* __restrict__ wlist,
                  const int* __restrict__ countPtr,
                  __bf16* __restrict__ aout, int Nld) {
  __shared__ __bf16 lds_a [32][LDSW];
  __shared__ __bf16 lds_wg[128][LDSW];
  __shared__ __bf16 lds_wu[128][LDSW];
  __shared__ int   s_tok[32];
  __shared__ float s_w[32];
  __shared__ int   s_count;

  const int tid = threadIdx.x;
  if (tid == 0) s_count = *countPtr;
  __syncthreads();
  const int mbase = blockIdx.x * 32;
  if (mbase >= s_count) return;  // block-uniform early exit

  if (tid < 32) {
    const int slot = mbase + tid;
    const bool ok  = slot < s_count;
    s_tok[tid] = ok ? tlist[slot] : 0;
    s_w[tid]   = ok ? wlist[slot] : 0.0f;  // pad slots contribute zero
  }
  __syncthreads();

  const int nbase = blockIdx.y * 128;
  const int lane  = tid & 31;
  const int wv    = tid >> 5;
  const int arow  = tid >> 3;        // 0..15 (also handles arow+16)
  const int acol  = (tid & 7) * 4;
  const int wn    = (tid & 31) * 4;  // W column group
  const int kq    = tid >> 5;        // W k-chunk [8kq, 8kq+8)

  // hoisted gathered-row pointers (constant across the K loop)
  const float* xr0 = x + (size_t)s_tok[arow]      * HDIM + acol;
  const float* xr1 = x + (size_t)s_tok[arow + 16] * HDIM + acol;

  v8f accg[2][2] = {};  // [m-subtile][n-half]
  v8f accu[2][2] = {};

  for (int kc = 0; kc < HDIM; kc += 32) {
    // ---- A: gathered 32x32 token tile, fp32 -> bf16, packed b64 stores ----
    {
      const float4 a0 = *reinterpret_cast<const float4*>(xr0 + kc);
      const float4 a1 = *reinterpret_cast<const float4*>(xr1 + kc);
      const v4bf p0 = {(__bf16)a0.x, (__bf16)a0.y, (__bf16)a0.z, (__bf16)a0.w};
      const v4bf p1 = {(__bf16)a1.x, (__bf16)a1.y, (__bf16)a1.z, (__bf16)a1.w};
      *reinterpret_cast<v4bf*>(&lds_a[arow][acol])      = p0;
      *reinterpret_cast<v4bf*>(&lds_a[arow + 16][acol]) = p1;
    }
    // ---- W tiles 32x128 -> bf16, LDS-transposed [n][k] ----
    load_w_tile(Wg + (size_t)kc * Nld + nbase, Nld, wn, kq, lds_wg);
    load_w_tile(Wu + (size_t)kc * Nld + nbase, Nld, wn, kq, lds_wu);
    if (kc + 32 < HDIM) {
      __builtin_prefetch(Wg + (size_t)(kc + 32) * Nld + nbase + wn, 0, 1);
      __builtin_prefetch(Wu + (size_t)(kc + 32) * Nld + nbase + wn, 0, 1);
    }
    __syncthreads();

    // ---- fragments ----
    const int am  = lane & 15;
    const int kb  = (lane >> 4) * 8;
    const v16bf af0 = frag16(&lds_a[am][kb],      &lds_a[am][kb + 16]);
    const v16bf af1 = frag16(&lds_a[16 + am][kb], &lds_a[16 + am][kb + 16]);

    const int bn  = lane & 15;
    const int kbb = (lane >> 4) * 8;
    {
      const v16bf b = frag16(&lds_wg[wv * 32 + bn][kbb], &lds_wg[wv * 32 + bn][kbb + 16]);
      accg[0][0] = bwmma(af0, b, accg[0][0]);
      accg[1][0] = bwmma(af1, b, accg[1][0]);
    }
    {
      const v16bf b = frag16(&lds_wg[wv * 32 + 16 + bn][kbb], &lds_wg[wv * 32 + 16 + bn][kbb + 16]);
      accg[0][1] = bwmma(af0, b, accg[0][1]);
      accg[1][1] = bwmma(af1, b, accg[1][1]);
    }
    {
      const v16bf b = frag16(&lds_wu[wv * 32 + bn][kbb], &lds_wu[wv * 32 + bn][kbb + 16]);
      accu[0][0] = bwmma(af0, b, accu[0][0]);
      accu[1][0] = bwmma(af1, b, accu[1][0]);
    }
    {
      const v16bf b = frag16(&lds_wu[wv * 32 + 16 + bn][kbb], &lds_wu[wv * 32 + 16 + bn][kbb + 16]);
      accu[0][1] = bwmma(af0, b, accu[0][1]);
      accu[1][1] = bwmma(af1, b, accu[1][1]);
    }
    __syncthreads();
  }

  // ---- epilogue: silu(g) * u * routing_weight -> bf16 scratch ----
#pragma unroll
  for (int mt = 0; mt < 2; ++mt) {
#pragma unroll
    for (int half = 0; half < 2; ++half) {
      const v8f cg = accg[mt][half];
      const v8f cu = accu[mt][half];
      const int n  = nbase + wv * 32 + half * 16 + (lane & 15);
#pragma unroll
      for (int r = 0; r < 8; ++r) {
        const int m = mt * 16 + r + 8 * (lane >> 4);
        const float g   = cg[r];
        const float u   = cu[r];
        const float act = (g / (1.0f + __expf(-g))) * u * s_w[m];
        aout[(size_t)(mbase + m) * Nld + n] = (__bf16)act;
      }
    }
  }
}

// ====================================================================
// Kernel 3: down GEMM. A = bf16 intermediate (direct per-lane global
// b128 loads), B = Wd[k][h] converted + LDS-transposed. Accumulates
// into out[t][h] with global_atomic_add_f32.
// ====================================================================
__global__ __launch_bounds__(128)
void moe_down_k(const __bf16* __restrict__ a, const float* __restrict__ Wd,
                const int* __restrict__ tlist, const int* __restrict__ countPtr,
                float* __restrict__ out, int Kd) {
  __shared__ __bf16 lds_w[128][LDSW];
  __shared__ int s_tok[32];
  __shared__ int s_count;

  const int tid = threadIdx.x;
  if (tid == 0) s_count = *countPtr;
  __syncthreads();
  const int mbase = blockIdx.x * 32;
  if (mbase >= s_count) return;

  if (tid < 32) {
    const int slot = mbase + tid;
    s_tok[tid] = (slot < s_count) ? tlist[slot] : 0;  // pad rows hold zeros
  }
  __syncthreads();

  const int hbase = blockIdx.y * 128;
  const int lane  = tid & 31;
  const int wv    = tid >> 5;
  const int wn    = (tid & 31) * 4;
  const int kq    = tid >> 5;

  v8f c[2][2] = {};  // [m-subtile][n-half]
  const __bf16* ar0 = a + (size_t)(mbase +      (lane & 15)) * Kd;
  const __bf16* ar1 = a + (size_t)(mbase + 16 + (lane & 15)) * Kd;
  const int kb = (lane >> 4) * 8;

  for (int kc = 0; kc < Kd; kc += 32) {
    load_w_tile(Wd + (size_t)kc * HDIM + hbase, HDIM, wn, kq, lds_w);
    if (kc + 32 < Kd)
      __builtin_prefetch(Wd + (size_t)(kc + 32) * HDIM + hbase + wn, 0, 1);
    __syncthreads();

    // A fragments straight from global bf16 (row-major K)
    const v16bf af0 = frag16(ar0 + kc + kb, ar0 + kc + kb + 16);
    const v16bf af1 = frag16(ar1 + kc + kb, ar1 + kc + kb + 16);

    const int bn  = lane & 15;
    const int kbb = (lane >> 4) * 8;
    {
      const v16bf b = frag16(&lds_w[wv * 32 + bn][kbb], &lds_w[wv * 32 + bn][kbb + 16]);
      c[0][0] = bwmma(af0, b, c[0][0]);
      c[1][0] = bwmma(af1, b, c[1][0]);
    }
    {
      const v16bf b = frag16(&lds_w[wv * 32 + 16 + bn][kbb], &lds_w[wv * 32 + 16 + bn][kbb + 16]);
      c[0][1] = bwmma(af0, b, c[0][1]);
      c[1][1] = bwmma(af1, b, c[1][1]);
    }
    __syncthreads();
  }

  // ---- epilogue: scatter-accumulate into out[t][h] ----
#pragma unroll
  for (int mt = 0; mt < 2; ++mt) {
#pragma unroll
    for (int r = 0; r < 8; ++r) {
      const int m = mt * 16 + r + 8 * (lane >> 4);
      const int t = s_tok[m];
      atomicAdd(&out[(size_t)t * HDIM + hbase + wv * 32 +      (lane & 15)], c[mt][0][r]);
      atomicAdd(&out[(size_t)t * HDIM + hbase + wv * 32 + 16 + (lane & 15)], c[mt][1][r]);
    }
  }
}

// ====================================================================
// Host-side launcher
// ====================================================================
extern "C" void kernel_launch(void* const* d_in, const int* in_sizes, int n_in,
                              void* d_out, int out_size, void* d_ws, size_t ws_size,
                              hipStream_t stream) {
  const float* x       = (const float*)d_in[0];
  const float* gate_w  = (const float*)d_in[1];
  const float* w_gate  = (const float*)d_in[2];
  const float* w_up    = (const float*)d_in[3];
  const float* w_down  = (const float*)d_in[4];
  const float* sw_gate = (const float*)d_in[5];
  const float* sw_up   = (const float*)d_in[6];
  const float* sw_down = (const float*)d_in[7];
  float* out = (float*)d_out;

  // workspace layout (~50 MB): counts | tlist | wlist | id-list | bf16 interm.
  char* ws = (char*)d_ws;
  const size_t OFF_TLIST = 256;
  const size_t OFF_WLIST = OFF_TLIST + (size_t)NEXP * TTOK * 4;
  const size_t OFF_STL   = OFF_WLIST + (size_t)NEXP * TTOK * 4;
  const size_t OFF_SWL   = OFF_STL + (size_t)TTOK * 4;
  const size_t OFF_A     = (OFF_SWL + (size_t)TTOK * 4 + 255) & ~(size_t)255;

  int*    counts  = (int*)ws;                    // NEXP + 1 (shared)
  int*    tlist   = (int*)(ws + OFF_TLIST);
  float*  wlist   = (float*)(ws + OFF_WLIST);
  int*    s_tlist = (int*)(ws + OFF_STL);
  float*  s_wlist = (float*)(ws + OFF_SWL);
  __bf16* aR      = (__bf16*)(ws + OFF_A);                 // [E][T][I]
  __bf16* aS      = aR + (size_t)NEXP * TTOK * IDIM;       // [T][IS]

  hipMemsetAsync(counts, 0, 256, stream);
  hipMemsetAsync(out, 0, (size_t)TTOK * HDIM * sizeof(float), stream);

  moe_router_k<<<TTOK / 8, 256, 0, stream>>>(x, gate_w, counts, tlist, wlist,
                                             s_tlist, s_wlist);

  // routed experts: per-expert launches keep each ~35 MB weight set L2-resident
  for (int e = 0; e < NEXP; ++e) {
    moe_gateup_k<<<dim3(TTOK / 32, IDIM / 128), 128, 0, stream>>>(
        x, w_gate + (size_t)e * HDIM * IDIM, w_up + (size_t)e * HDIM * IDIM,
        tlist + e * TTOK, wlist + e * TTOK, counts + e,
        aR + (size_t)e * TTOK * IDIM, IDIM);
  }
  // shared expert (identity routing, weight 1.0)
  moe_gateup_k<<<dim3(TTOK / 32, ISDIM / 128), 128, 0, stream>>>(
      x, sw_gate, sw_up, s_tlist, s_wlist, counts + NEXP, aS, ISDIM);

  for (int e = 0; e < NEXP; ++e) {
    moe_down_k<<<dim3(TTOK / 32, HDIM / 128), 128, 0, stream>>>(
        aR + (size_t)e * TTOK * IDIM, w_down + (size_t)e * IDIM * HDIM,
        tlist + e * TTOK, counts + e, out, IDIM);
  }
  moe_down_k<<<dim3(TTOK / 32, HDIM / 128), 128, 0, stream>>>(
      aS, sw_down, s_tlist, counts + NEXP, out, ISDIM);
}
